// CNN_Decoder_77283641524936
// MI455X (gfx1250) — compile-verified, matrix operand order checked by reading, not run on hardware
//
#include <hip/hip_runtime.h>
#include <hip/hip_bf16.h>

// ---------------------------------------------------------------------------
// ConvS2S decoder for MI455X (gfx1250).
// All GEMMs -> v_wmma_f32_16x16x32_bf16 (fp32 in HBM, bf16 in LDS, f32 acc).
// Compute-bound (~590 GFLOP vs ~320 MB @ 23.3 TB/s), so: branchless vectorized
// global->LDS staging, register-pipelined prefetch, 16 WMMA per stage.
// ---------------------------------------------------------------------------

typedef __attribute__((ext_vector_type(16))) __bf16 v16bf;
typedef __attribute__((ext_vector_type(8)))  __bf16 v8bf;
typedef __attribute__((ext_vector_type(4)))  __bf16 v4bf;
typedef __attribute__((ext_vector_type(8)))  float  v8f;
typedef __attribute__((ext_vector_type(4)))  float  v4f;

#define BMT 128
#define BNT 128
#define BKT 64
#define LDT 72   // BKT + 8 bf16 pad -> 144B row stride: 16B aligned, 36-bank
                 // rotation per row => ds_load_b128 frag reads are conflict-free

union FragAB { v16bf v; v8bf h[2]; };

// Batched GEMM: C[m,n] = sum_k A'[m,k] * B'[n,k]
//   A'(r,k) = A[(r+a_shift)*lda + k], valid iff (r%a_period)+a_shift >= 0
//   B'(n,k) = Bp[n*b_nstride + k]            (k-contiguous, always)
// Requirements (guaranteed by host): K%4==0, lda%4==0, b_nstride%4==0.
// Epilogue: out = (acc + bias[n] + add_src[r,n]) * scale  (+ C_old if acc_flag)
__global__ __launch_bounds__(256)
void gemm_wmma_bf16(const float* __restrict__ A, long lda, long batchA,
                    int a_shift, int a_period,
                    const float* __restrict__ Bp, long b_nstride, long batchB,
                    float* __restrict__ C, long ldc, long batchC,
                    const float* __restrict__ bias,
                    const float* __restrict__ add_src,
                    float scale, int acc_flag,
                    int M, int N, int K)
{
    const int tid  = threadIdx.x;
    const int lane = tid & 31;
    const int wid  = tid >> 5;
    const int m0   = blockIdx.y * BMT;
    const int n0   = blockIdx.x * BNT;
    const long bz  = blockIdx.z;

    A  += bz * batchA;
    Bp += bz * batchB;
    C  += bz * batchC;
    const float* add = add_src ? (add_src + bz * batchC) : nullptr;

    __shared__ __align__(16) __bf16 As[BMT * LDT];
    __shared__ __align__(16) __bf16 Bs[BNT * LDT];

    v8f acc[4][2];
    const v8f vzero = {0.f, 0.f, 0.f, 0.f, 0.f, 0.f, 0.f, 0.f};
#pragma unroll
    for (int i = 0; i < 4; ++i)
#pragma unroll
        for (int j = 0; j < 2; ++j) acc[i][j] = vzero;

    const int wm0   = (wid >> 2) * 64;   // 2 wave-rows x 4 wave-cols
    const int wn0   = (wid & 3) * 32;
    const int lm    = lane & 15;
    const int khalf = (lane >> 4) * 8;   // ISA 16-bit layout: hi-lanes hold K 8-15/24-31

    // register staging: 8 float4 chunks per matrix per thread (128x64 tile)
    v4f ra[8], rb[8];

    auto fetch = [&](int k0) {
#pragma unroll
        for (int t = 0; t < 8; ++t) {
            int id  = tid + 256 * t;           // 0..2047 float4-chunks
            int row = id >> 4;
            int col = (id & 15) << 2;
            // ---- A chunk (branchless: clamp offset, load, select-zero) ----
            {
                int r = m0 + row, k = k0 + col;
                bool v = (r < M) && (k < K);
                if (a_shift != 0)              // uniform branch (SGPR arg)
                    v = v && ((r % a_period) + a_shift >= 0);
                long off = v ? ((long)(r + a_shift) * lda + k) : 0;
                v4f x = *(const v4f*)(A + off);
#pragma unroll
                for (int c = 0; c < 4; ++c) ra[t][c] = v ? x[c] : 0.f;
            }
            // ---- B chunk ----
            {
                int n = n0 + row, k = k0 + col;
                bool v = (n < N) && (k < K);
                long off = v ? ((long)n * b_nstride + k) : 0;
                v4f x = *(const v4f*)(Bp + off);
#pragma unroll
                for (int c = 0; c < 4; ++c) rb[t][c] = v ? x[c] : 0.f;
            }
        }
    };

    fetch(0);

    for (int k0 = 0; k0 < K; k0 += BKT) {
        __syncthreads();   // prior compute done reading LDS
        // ---- regs -> LDS (fp32 -> bf16) ----
#pragma unroll
        for (int t = 0; t < 8; ++t) {
            int id  = tid + 256 * t;
            int row = id >> 4;
            int col = (id & 15) << 2;
            v4bf ha, hb;
#pragma unroll
            for (int c = 0; c < 4; ++c) {
                ha[c] = (__bf16)ra[t][c];
                hb[c] = (__bf16)rb[t][c];
            }
            *(v4bf*)&As[row * LDT + col] = ha;
            *(v4bf*)&Bs[row * LDT + col] = hb;
        }
        __syncthreads();

        // prefetch next tile while WMMAs run
        if (k0 + BKT < K) fetch(k0 + BKT);

        // ---- compute: 2 k-slices of 32, 8 WMMA each ----
#pragma unroll
        for (int ks = 0; ks < BKT; ks += 32) {
            FragAB a[4], b[2];
#pragma unroll
            for (int i = 0; i < 4; ++i) {
                int row = wm0 + 16 * i + lm;
                a[i].h[0] = *(const v8bf*)&As[row * LDT + ks + khalf];
                a[i].h[1] = *(const v8bf*)&As[row * LDT + ks + 16 + khalf];
            }
#pragma unroll
            for (int j = 0; j < 2; ++j) {
                int nn = wn0 + 16 * j + lm;
                b[j].h[0] = *(const v8bf*)&Bs[nn * LDT + ks + khalf];
                b[j].h[1] = *(const v8bf*)&Bs[nn * LDT + ks + 16 + khalf];
            }
#pragma unroll
            for (int i = 0; i < 4; ++i)
#pragma unroll
                for (int j = 0; j < 2; ++j)
                    acc[i][j] = __builtin_amdgcn_wmma_f32_16x16x32_bf16(
                        false, a[i].v, false, b[j].v, (short)0, acc[i][j], false, false);
        }
    }

    // ---- epilogue: C/D frag layout: VGPR e -> row e + 8*(lane>=16), col lane&15
    const int rsel = (lane >> 4) << 3;
#pragma unroll
    for (int i = 0; i < 4; ++i)
#pragma unroll
        for (int j = 0; j < 2; ++j)
#pragma unroll
            for (int e = 0; e < 8; ++e) {
                int r = m0 + wm0 + 16 * i + e + rsel;
                int c = n0 + wn0 + 16 * j + lm;
                if (r < M && c < N) {
                    float v = acc[i][j][e];
                    if (bias) v += bias[c];
                    if (add)  v += add[(long)r * ldc + c];
                    v *= scale;
                    long off = (long)r * ldc + c;
                    if (acc_flag) v += C[off];
                    C[off] = v;
                }
            }
}

// embedded[b,t,:] = src_emb[target[b,t],:] + pos_emb[t,:]
__global__ void embed_kernel(const int* __restrict__ tgt,
                             const float* __restrict__ src_emb,
                             const float* __restrict__ pos_emb,
                             float* __restrict__ out, long total, int T, int E)
{
    long i = (long)blockIdx.x * blockDim.x + threadIdx.x;
    if (i >= total) return;
    long r = i / E; int e = (int)(i % E);
    int t = (int)(r % T);
    int tok = tgt[r];
    out[i] = src_emb[(long)tok * E + e] + pos_emb[(long)t * E + e];
}

// GLU: out[r,h] = c[r,h] * sigmoid(c[r,H+h]),  c is [rows, 2H]
__global__ void glu_kernel(const float* __restrict__ c, float* __restrict__ out,
                           long total, int H)
{
    long i = (long)blockIdx.x * blockDim.x + threadIdx.x;
    if (i >= total) return;
    long r = i / H; int h = (int)(i % H);
    float a = c[r * (2 * H) + h];
    float g = c[r * (2 * H) + H + h];
    out[i] = a / (1.f + __expf(-g));
}

// ci[i] = (c2[i] + ci[i]) * scale
__global__ void residual_kernel(const float* __restrict__ c2, float* __restrict__ ci,
                                long total, float scale)
{
    long i = (long)blockIdx.x * blockDim.x + threadIdx.x;
    if (i >= total) return;
    ci[i] = (c2[i] + ci[i]) * scale;
}

// row-wise softmax, one wave32 per row, S <= 128
__global__ void softmax_kernel(const float* __restrict__ x, float* __restrict__ y,
                               int S)
{
    int r = blockIdx.x;
    int lane = threadIdx.x;
    float vals[4];
    float mx = -INFINITY;
#pragma unroll
    for (int i = 0; i < 4; ++i) {
        int s = lane + 32 * i;
        float v = (s < S) ? x[(long)r * S + s] : -INFINITY;
        vals[i] = v;
        mx = fmaxf(mx, v);
    }
#pragma unroll
    for (int o = 16; o > 0; o >>= 1) mx = fmaxf(mx, __shfl_xor(mx, o, 32));
    float sum = 0.f;
#pragma unroll
    for (int i = 0; i < 4; ++i) {
        int s = lane + 32 * i;
        float e = (s < S) ? __expf(vals[i] - mx) : 0.f;
        vals[i] = e;
        sum += e;
    }
#pragma unroll
    for (int o = 16; o > 0; o >>= 1) sum += __shfl_xor(sum, o, 32);
    float inv = 1.f / sum;
#pragma unroll
    for (int i = 0; i < 4; ++i) {
        int s = lane + 32 * i;
        if (s < S) y[(long)r * S + s] = vals[i] * inv;
    }
}

// wpack[k][o][i] = conv_w_layer[o][i][k]   (makes conv B-operand k-contiguous)
__global__ void conv_repack_kernel(const float* __restrict__ w, float* __restrict__ out,
                                   long total, int H2, int H)
{
    long i = (long)blockIdx.x * blockDim.x + threadIdx.x;
    if (i >= total) return;
    long per = (long)H2 * H;
    int k = (int)(i / per);
    long rem = i % per;
    long o = rem / H;
    int ii = (int)(rem % H);
    out[i] = w[o * (long)H * 3 + (long)ii * 3 + k];
}

// encT[b][e][s] = enc[b][s][e]   (makes attention B-operand k-contiguous)
__global__ void transpose_enc_kernel(const float* __restrict__ enc, float* __restrict__ out,
                                     long total, int S, int E)
{
    long i = (long)blockIdx.x * blockDim.x + threadIdx.x;
    if (i >= total) return;
    long per = (long)S * E;
    long b = i / per;
    long rem = i % per;
    long e = rem / S;
    long s = rem % S;
    out[i] = enc[b * per + s * E + e];
}

// ---------------------------------------------------------------------------

static inline void launch_gemm(hipStream_t stream,
                               const float* A, long lda, long batchA, int a_shift, int a_period,
                               const float* B, long bn, long batchB,
                               float* C, long ldc, long batchC,
                               const float* bias, const float* add,
                               float scale, int accf,
                               int M, int N, int K, int nbatch)
{
    dim3 grid((N + BNT - 1) / BNT, (M + BMT - 1) / BMT, nbatch);
    gemm_wmma_bf16<<<grid, 256, 0, stream>>>(A, lda, batchA, a_shift, a_period,
                                             B, bn, batchB,
                                             C, ldc, batchC,
                                             bias, add, scale, accf, M, N, K);
}

extern "C" void kernel_launch(void* const* d_in, const int* in_sizes, int n_in,
                              void* d_out, int out_size, void* d_ws, size_t ws_size,
                              hipStream_t stream)
{
    (void)in_sizes; (void)n_in; (void)out_size; (void)ws_size;

    const int V = 32000, E = 512, H = 1024, L = 10;
    const int B = 32, T = 100, S = 100;
    const long BT = (long)B * T;                  // 3200
    const float scale = 0.7071067811865476f;      // sqrt(0.5)

    const int*   target       = (const int*)  d_in[0];
    const float* enc_conved   = (const float*)d_in[1];   // [B,S,E]
    const float* enc_combined = (const float*)d_in[2];   // [B,S,E]
    const float* src_emb      = (const float*)d_in[3];   // [V,E]
    const float* pos_emb      = (const float*)d_in[4];   // [120,E]
    const float* w_eh         = (const float*)d_in[5];   // [H,E]
    const float* b_eh         = (const float*)d_in[6];
    const float* w_he         = (const float*)d_in[7];   // [E,H]
    const float* b_he         = (const float*)d_in[8];
    const float* w_ahe        = (const float*)d_in[9];   // [E,H]
    const float* b_ahe        = (const float*)d_in[10];
    const float* w_aeh        = (const float*)d_in[11];  // [H,E]
    const float* b_aeh        = (const float*)d_in[12];
    const float* w_fc         = (const float*)d_in[13];  // [V,E]
    const float* b_fc         = (const float*)d_in[14];
    const float* conv_w       = (const float*)d_in[15];  // [L,2H,H,3]
    const float* conv_b       = (const float*)d_in[16];  // [L,2H]

    float* logits  = (float*)d_out;                      // [B,T,V]
    float* att_out = logits + BT * (long)V;              // [B,T,S]

    // workspace (fp32)
    float* p = (float*)d_ws;
    float* embedded = p; p += BT * E;            // [BT,E]
    float* ci       = p; p += BT * H;            // conv_input, [BT,H]
    float* cbuf     = p; p += BT * 2 * H;        // conv output, [BT,2H]
    float* conved   = p; p += BT * H;            // [BT,H]
    float* comb     = p; p += BT * E;            // [BT,E]
    float* energy   = p; p += (long)B * T * S;
    float* att_enc  = p; p += BT * E;            // [BT,E]
    float* conved2  = p; p += BT * H;            // [BT,H]
    float* out_e    = p; p += BT * E;            // [BT,E]
    float* encT     = p; p += (long)B * E * S;   // enc_combined^T, [B,E,S]
    float* wpack    = p; p += (long)3 * 2 * H * H; // per-layer conv weights [3][2H][H]

    // 0) one-time transpose of encoder_combined -> [B,E,S]
    {
        long total = (long)B * E * S;
        transpose_enc_kernel<<<(unsigned)((total + 255) / 256), 256, 0, stream>>>(
            enc_combined, encT, total, S, E);
    }

    // 1) embedding
    {
        long total = BT * E;
        embed_kernel<<<(unsigned)((total + 255) / 256), 256, 0, stream>>>(
            target, src_emb, pos_emb, embedded, total, T, E);
    }

    // 2) input projection: ci = embedded @ w_eh^T + b_eh   [BT,H]
    launch_gemm(stream, embedded, E, 0, 0, 1,
                w_eh, E, 0,
                ci, H, 0, b_eh, nullptr, 1.f, 0,
                (int)BT, H, E, 1);

    for (int l = 0; l < L; ++l) {
        const float* cw = conv_w + (long)l * 2 * H * H * 3;
        const float* cb = conv_b + (long)l * 2 * H;

        // 3a) repack layer weights: [2H,H,3] -> [3][2H][H]
        {
            long total = (long)3 * 2 * H * H;
            conv_repack_kernel<<<(unsigned)((total + 255) / 256), 256, 0, stream>>>(
                cw, wpack, total, 2 * H, H);
        }

        // 3b) causal conv k=3 as 3 accumulated GEMMs over shifted inputs
        for (int k = 0; k < 3; ++k) {
            launch_gemm(stream, ci, H, 0, /*a_shift*/ k - 2, /*a_period*/ T,
                        wpack + (long)k * 2 * H * H, H, 0,
                        cbuf, 2 * H, 0,
                        (k == 0) ? cb : nullptr, nullptr, 1.f, /*acc*/ (k > 0),
                        (int)BT, 2 * H, H, 1);
        }

        // 4) GLU
        {
            long total = BT * H;
            glu_kernel<<<(unsigned)((total + 255) / 256), 256, 0, stream>>>(
                cbuf, conved, total, H);
        }

        // 5) comb = (conved @ w_ahe^T + b_ahe + embedded) * scale   [BT,E]
        launch_gemm(stream, conved, H, 0, 0, 1,
                    w_ahe, H, 0,
                    comb, E, 0, b_ahe, embedded, scale, 0,
                    (int)BT, E, H, 1);

        // 6) energy[b] = comb[b] @ enc_conved[b]^T   [T,S], batched over B
        launch_gemm(stream, comb, E, (long)T * E, 0, 1,
                    enc_conved, E, (long)S * E,
                    energy, S, (long)T * S, nullptr, nullptr, 1.f, 0,
                    T, S, E, B);

        // 7) softmax -> attention (written straight into its d_out slice)
        softmax_kernel<<<(unsigned)BT, 32, 0, stream>>>(energy, att_out, S);

        // 8) att_enc[b] = attention[b] @ enc_combined[b]   [T,E], batched
        //    B-operand = encT[b] : B'(n=e, k=s) = encT[b][e][s], nstride=S
        launch_gemm(stream, att_out, S, (long)T * S, 0, 1,
                    encT, S, (long)E * S,
                    att_enc, E, (long)T * E, nullptr, nullptr, 1.f, 0,
                    T, E, S, B);

        // 9) conved2 = (att_enc @ w_aeh^T + b_aeh + conved) * scale   [BT,H]
        launch_gemm(stream, att_enc, E, 0, 0, 1,
                    w_aeh, E, 0,
                    conved2, H, 0, b_aeh, conved, scale, 0,
                    (int)BT, H, E, 1);

        // 10) ci = (conved2 + ci) * scale
        {
            long total = BT * H;
            residual_kernel<<<(unsigned)((total + 255) / 256), 256, 0, stream>>>(
                conved2, ci, total, scale);
        }
    }

    // 11) out_e = ci @ w_he^T + b_he   [BT,E]
    launch_gemm(stream, ci, H, 0, 0, 1,
                w_he, H, 0,
                out_e, E, 0, b_he, nullptr, 1.f, 0,
                (int)BT, E, H, 1);

    // 12) logits = out_e @ w_fc^T + b_fc   [BT,V]
    launch_gemm(stream, out_e, E, 0, 0, 1,
                w_fc, E, 0,
                logits, V, 0, b_fc, nullptr, 1.f, 0,
                (int)BT, V, E, 1);
}
